// MultiHeadAttention_25074019074053
// MI455X (gfx1250) — matrix operand order repaired and checked by reading, not compiled, not run on hardware
//
#include <hip/hip_runtime.h>
#include <hip/hip_bf16.h>

// MHA forward for gfx1250 (MI455X), bf16 WMMA w/ fp32 accumulate.
// B=2, S=2048, D=1024, H=16, DK=64. Outputs: out [B,S,D] fp32 then attn [B,H,S,S] fp32.

#define BB 2
#define SS 2048
#define DD 1024
#define HH 16
#define DKK 64
#define SPAD 2052   // padded LDS row stride (floats): %4==0 for b128, 8*SPAD%64==32

typedef __attribute__((ext_vector_type(16))) __bf16 v16bf;
typedef __attribute__((ext_vector_type(8)))  __bf16 v8bf;
typedef __attribute__((ext_vector_type(8)))  float  v8f;
typedef unsigned short bf16_t;

__device__ __forceinline__ unsigned short f2bf_u(float f) {
  unsigned int u = __builtin_bit_cast(unsigned int, f);
  unsigned int r = u + 0x7FFFu + ((u >> 16) & 1u);   // round-to-nearest-even
  return (unsigned short)(r >> 16);
}
__device__ __forceinline__ __bf16 f2bf(float f) {
  unsigned short h = f2bf_u(f);
  return __builtin_bit_cast(__bf16, h);
}

__device__ __forceinline__ v8f wmma_bf16(v16bf a, v16bf b, v8f c) {
  return __builtin_amdgcn_wmma_f32_16x16x32_bf16(false, a, false, b, (short)0, c,
                                                 false, false);
}

// Fill v[off..off+7] from 8 contiguous fp32 (two float4 = 2x global_load_b128).
__device__ __forceinline__ void load8f_bf(const float* p, v16bf& v, int off) {
  float4 a = reinterpret_cast<const float4*>(p)[0];
  float4 b = reinterpret_cast<const float4*>(p)[1];
  v[off + 0] = f2bf(a.x); v[off + 1] = f2bf(a.y);
  v[off + 2] = f2bf(a.z); v[off + 3] = f2bf(a.w);
  v[off + 4] = f2bf(b.x); v[off + 5] = f2bf(b.y);
  v[off + 6] = f2bf(b.z); v[off + 7] = f2bf(b.w);
}

// ---------------------------------------------------------------------------
// Projection GEMM: out[m,n] = sum_k X[m,k]*W[n,k] + bias[n]
// TRANSPOSE_OUT=false -> store bf16 at [bh, s, dk]   (for Q, K)
// TRANSPOSE_OUT=true  -> store bf16 at [bh, dk, s]   (for V, B-operand-ready)
// One wave computes a 16(M) x 64(N) output strip. 4 waves per block.
// ---------------------------------------------------------------------------
template <bool TRANSPOSE_OUT>
__global__ __launch_bounds__(128) void proj_kernel(const float* __restrict__ X,
                                                   const float* __restrict__ W,
                                                   const float* __restrict__ bias,
                                                   bf16_t* __restrict__ out) {
  const int wave = threadIdx.x >> 5;
  const int lane = threadIdx.x & 31;
  const int job  = blockIdx.x * 4 + wave;   // 4096 jobs
  const int m0   = (job >> 4) * 16;         // 256 M tiles
  const int n0   = (job & 15) * 64;         // 16 N groups of 64
  const int lm   = lane & 15;
  const int hi   = lane >> 4;

  v8f acc[4];
#pragma unroll
  for (int i = 0; i < 4; ++i) acc[i] = (v8f)0.0f;

  const float* arowb = X + (size_t)(m0 + lm) * DD;
  for (int k0 = 0; k0 < DD; k0 += 32) {
    // A fragment: X[m0+lm, k0 + e + 16*hi]
    v16bf a;
    load8f_bf(arowb + k0 + 16 * hi, a, 0);
    load8f_bf(arowb + k0 + 16 * hi + 8, a, 8);
#pragma unroll
    for (int sub = 0; sub < 4; ++sub) {
      // B fragment: W[n0+sub*16+lm, k0 + e + 16*hi]  (16 contiguous fp32)
      const float* brow = W + (size_t)(n0 + sub * 16 + lm) * DD + k0 + 16 * hi;
      v16bf b;
      load8f_bf(brow, b, 0);
      load8f_bf(brow + 8, b, 8);
      acc[sub] = wmma_bf16(a, b, acc[sub]);
    }
  }

#pragma unroll
  for (int sub = 0; sub < 4; ++sub) {
#pragma unroll
    for (int r = 0; r < 8; ++r) {
      const int m = m0 + r + 8 * hi;         // global row in [0, B*S)
      const int n = n0 + sub * 16 + lm;      // global col in [0, D)
      const float v = acc[sub][r] + bias[n];
      const int b  = m >> 11;                // / S
      const int s  = m & (SS - 1);
      const int h  = n >> 6;                 // / DK
      const int dk = n & (DKK - 1);
      size_t idx;
      if (TRANSPOSE_OUT)
        idx = (((size_t)(b * HH + h)) * DKK + dk) * SS + s;
      else
        idx = (((size_t)(b * HH + h)) * SS + s) * DKK + dk;
      out[idx] = f2bf_u(v);
    }
  }
}

// ---------------------------------------------------------------------------
// Fused attention: one workgroup (8 waves, 256 thr) per (b, h, 16-query tile).
// LDS: scores[16][SPAD] fp32 (~129 KB) + red[256] fp32, dynamic.
// Wave w owns keys [w*256, w*256+256).
// Mask is applied (vectorized, row-major) inside softmax pass 1, not in the
// scores phase, so the 537 MB mask stream is read as int4 instead of scattered
// b32 columns.
// ---------------------------------------------------------------------------
__global__ __launch_bounds__(256) void attn_kernel(const bf16_t* __restrict__ Qb,
                                                   const bf16_t* __restrict__ Kb,
                                                   const bf16_t* __restrict__ Vt,
                                                   const int* __restrict__ mask,
                                                   float* __restrict__ attn_out,
                                                   float* __restrict__ ctx) {
  extern __shared__ float smem[];
  float* sc  = smem;              // 16 * SPAD
  float* red = smem + 16 * SPAD;  // 256

  const int tid  = threadIdx.x;
  const int wave = tid >> 5;
  const int lane = tid & 31;
  const int lm   = lane & 15;
  const int hi   = lane >> 4;

  const int bh = blockIdx.x >> 7;       // 128 q-tiles per (b,h)
  const int qt = blockIdx.x & 127;
  const int b  = bh >> 4;
  const int h  = bh & 15;
  const int q0 = qt * 16;

  const size_t qkBase = (size_t)bh * SS * DKK;  // [bh, s, dk]
  const size_t vtBase = (size_t)bh * DKK * SS;  // [bh, dk, s]

  // ---- Q fragments for this tile: aq[st] covers d = 32*st .. 32*st+31 ----
  v16bf aq[2];
  {
    const bf16_t* qrow = Qb + qkBase + (size_t)(q0 + lm) * DKK;
#pragma unroll
    for (int st = 0; st < 2; ++st) {
      const v8bf* p0 = reinterpret_cast<const v8bf*>(qrow + 32 * st + 16 * hi);
      v8bf lo = p0[0];          // d offsets 16*hi .. 16*hi+7
      v8bf hi8 = p0[1];         // d offsets 16*hi+8 .. 16*hi+15
#pragma unroll
      for (int e = 0; e < 8; ++e) { aq[st][e] = lo[e]; aq[st][8 + e] = hi8[e]; }
    }
  }

  // ---- scores = (Q K^T) * scale into LDS (mask applied later) ----
  const int kb0 = wave * 256;
  for (int kt = 0; kt < 16; ++kt) {
    const int kbase = kb0 + kt * 16;
    v8f c = (v8f)0.0f;
#pragma unroll
    for (int st = 0; st < 2; ++st) {
      // B operand: K[key = kbase+lm, d = 32*st + e + 16*hi] (16 contiguous bf16)
      const bf16_t* krow =
          Kb + qkBase + (size_t)(kbase + lm) * DKK + 32 * st + 16 * hi;
      v16bf bk = *reinterpret_cast<const v16bf*>(krow);
      c = wmma_bf16(aq[st], bk, c);
    }
    const int key = kbase + lm;
#pragma unroll
    for (int r = 0; r < 8; ++r) {
      const int q = r + 8 * hi;
      sc[q * SPAD + key] = c[r] * 0.125f;   // 1/sqrt(64)
    }
  }
  __syncthreads();

  // ---- softmax over rows (16 threads per row, float4-vectorized) ----
  {
    const int row = tid >> 4;        // 0..15
    const int c4  = (tid & 15) * 4;  // float4 column start; group covers 64 cols
    float* srow_p = sc + row * SPAD;
    const int* mrow_p = mask + ((size_t)b * SS + (q0 + row)) * SS;

    // pass 1: apply mask (int4 row-major loads) + running max
    float mx = -3.4e38f;
    for (int cix = c4; cix < SS; cix += 64) {
      float4 v = *reinterpret_cast<float4*>(srow_p + cix);
      int4 mk = *reinterpret_cast<const int4*>(mrow_p + cix);
      v.x = (mk.x == 0) ? -1e9f : v.x;
      v.y = (mk.y == 0) ? -1e9f : v.y;
      v.z = (mk.z == 0) ? -1e9f : v.z;
      v.w = (mk.w == 0) ? -1e9f : v.w;
      *reinterpret_cast<float4*>(srow_p + cix) = v;
      mx = fmaxf(mx, fmaxf(fmaxf(v.x, v.y), fmaxf(v.z, v.w)));
    }
    red[tid] = mx;
    __syncthreads();
    float mrow = red[row * 16];
#pragma unroll
    for (int i = 1; i < 16; ++i) mrow = fmaxf(mrow, red[row * 16 + i]);

    // pass 2: exp + running sum
    float s = 0.0f;
    for (int cix = c4; cix < SS; cix += 64) {
      float4 v = *reinterpret_cast<float4*>(srow_p + cix);
      v.x = __expf(v.x - mrow);
      v.y = __expf(v.y - mrow);
      v.z = __expf(v.z - mrow);
      v.w = __expf(v.w - mrow);
      *reinterpret_cast<float4*>(srow_p + cix) = v;
      s += v.x + v.y + v.z + v.w;
    }
    __syncthreads();
    red[tid] = s;
    __syncthreads();
    float srow = 0.0f;
#pragma unroll
    for (int i = 0; i < 16; ++i) srow += red[row * 16 + i];
    const float inv = 1.0f / srow;

    // pass 3: normalize in LDS + b128 stores of attn
    float* arow = attn_out + ((size_t)bh * SS + (q0 + row)) * SS;
    for (int cix = c4; cix < SS; cix += 64) {
      float4 v = *reinterpret_cast<float4*>(srow_p + cix);
      v.x *= inv; v.y *= inv; v.z *= inv; v.w *= inv;
      *reinterpret_cast<float4*>(srow_p + cix) = v;  // keep probs for A*V
      *reinterpret_cast<float4*>(arow + cix) = v;    // the 537 MB stream
    }
  }
  __syncthreads();

  // ---- ctx_partial = P * V over this wave's 256 keys (8 k-steps of 32) ----
  v8f cc[4];
#pragma unroll
  for (int i = 0; i < 4; ++i) cc[i] = (v8f)0.0f;

  for (int ks = 0; ks < 8; ++ks) {
    const int kbase = kb0 + ks * 32;
    // A operand: P[q = lm, k = kbase + e + 16*hi] from LDS
    v16bf ap;
#pragma unroll
    for (int e = 0; e < 16; ++e) {
      const int kk = kbase + e + 16 * hi;
      ap[e] = f2bf(sc[lm * SPAD + kk]);
    }
#pragma unroll
    for (int sub = 0; sub < 4; ++sub) {
      // B operand: Vt[d = sub*16+lm, s = kbase + e + 16*hi] (16 contiguous bf16)
      const bf16_t* vrow =
          Vt + vtBase + (size_t)(sub * 16 + lm) * SS + kbase + 16 * hi;
      v16bf bv = *reinterpret_cast<const v16bf*>(vrow);
      cc[sub] = wmma_bf16(ap, bv, cc[sub]);
    }
  }
  __syncthreads();  // all waves done reading probabilities from LDS

  // ---- reduce 8 wave-partials via LDS (reuse sc region, 8*16*64 floats) ----
#pragma unroll
  for (int sub = 0; sub < 4; ++sub)
#pragma unroll
    for (int r = 0; r < 8; ++r) {
      const int q = r + 8 * hi;
      const int d = sub * 16 + lm;
      sc[wave * 1024 + q * 64 + d] = cc[sub][r];
    }
  __syncthreads();

  for (int i = tid; i < 1024; i += 256) {
    float s = 0.0f;
#pragma unroll
    for (int w = 0; w < 8; ++w) s += sc[w * 1024 + i];
    const int q = i >> 6;
    const int d = i & 63;
    ctx[((size_t)b * SS + (q0 + q)) * DD + h * 64 + d] = s;
  }
}

// ---------------------------------------------------------------------------
// Output projection: out[m,n] = sum_k ctx[m,k]*Wo[n,k] + bo[n], fp32 out.
// ---------------------------------------------------------------------------
__global__ __launch_bounds__(128) void oproj_kernel(const float* __restrict__ ctx,
                                                    const float* __restrict__ Wo,
                                                    const float* __restrict__ bo,
                                                    float* __restrict__ out) {
  const int wave = threadIdx.x >> 5;
  const int lane = threadIdx.x & 31;
  const int job  = blockIdx.x * 4 + wave;
  const int m0   = (job >> 4) * 16;
  const int n0   = (job & 15) * 64;
  const int lm   = lane & 15;
  const int hi   = lane >> 4;

  v8f acc[4];
#pragma unroll
  for (int i = 0; i < 4; ++i) acc[i] = (v8f)0.0f;

  const float* arowb = ctx + (size_t)(m0 + lm) * DD;
  for (int k0 = 0; k0 < DD; k0 += 32) {
    v16bf a;
    load8f_bf(arowb + k0 + 16 * hi, a, 0);
    load8f_bf(arowb + k0 + 16 * hi + 8, a, 8);
#pragma unroll
    for (int sub = 0; sub < 4; ++sub) {
      const float* brow = Wo + (size_t)(n0 + sub * 16 + lm) * DD + k0 + 16 * hi;
      v16bf b;
      load8f_bf(brow, b, 0);
      load8f_bf(brow + 8, b, 8);
      acc[sub] = wmma_bf16(a, b, acc[sub]);
    }
  }

#pragma unroll
  for (int sub = 0; sub < 4; ++sub)
#pragma unroll
    for (int r = 0; r < 8; ++r) {
      const int m = m0 + r + 8 * hi;
      const int n = n0 + sub * 16 + lm;
      out[(size_t)m * DD + n] = acc[sub][r] + bo[n];
    }
}

// ---------------------------------------------------------------------------
extern "C" void kernel_launch(void* const* d_in, const int* in_sizes, int n_in,
                              void* d_out, int out_size, void* d_ws, size_t ws_size,
                              hipStream_t stream) {
  (void)in_sizes; (void)n_in; (void)out_size; (void)ws_size;
  const float* query = (const float*)d_in[0];
  const float* key_i = (const float*)d_in[1];
  const float* value = (const float*)d_in[2];
  const int*   mask  = (const int*)d_in[3];
  const float* Wq = (const float*)d_in[4];
  const float* bq = (const float*)d_in[5];
  const float* Wk = (const float*)d_in[6];
  const float* bk = (const float*)d_in[7];
  const float* Wv = (const float*)d_in[8];
  const float* bv = (const float*)d_in[9];
  const float* Wo = (const float*)d_in[10];
  const float* bo = (const float*)d_in[11];

  float* out  = (float*)d_out;                       // [B,S,D]
  float* attn = out + (size_t)BB * SS * DD;          // [B,H,S,S]

  const size_t nQ = (size_t)BB * HH * SS * DKK;      // 4,194,304
  bf16_t* Qb = (bf16_t*)d_ws;
  bf16_t* Kb = Qb + nQ;
  bf16_t* Vt = Kb + nQ;
  float*  ctx = (float*)(Vt + nQ);                   // [B,S,D] fp32

  dim3 pgrid(1024), pblock(128);
  proj_kernel<false><<<pgrid, pblock, 0, stream>>>(query, Wq, bq, Qb);
  proj_kernel<false><<<pgrid, pblock, 0, stream>>>(key_i, Wk, bk, Kb);
  proj_kernel<true ><<<pgrid, pblock, 0, stream>>>(value, Wv, bv, Vt);

  const size_t shmem = (size_t)16 * SPAD * sizeof(float) + 256 * sizeof(float);
  attn_kernel<<<dim3(BB * HH * (SS / 16)), dim3(256), shmem, stream>>>(
      Qb, Kb, Vt, mask, attn, ctx);

  oproj_kernel<<<pgrid, pblock, 0, stream>>>(ctx, Wo, bo, out);
}
